// BiAttention_6665789243995
// MI455X (gfx1250) — matrix operand order repaired and checked by reading, not compiled, not run on hardware
//
#include <hip/hip_runtime.h>
#include <hip/hip_bf16.h>
#include <math.h>

typedef __attribute__((ext_vector_type(16))) __bf16 v16bf;
typedef __attribute__((ext_vector_type(8)))  __bf16 v8bf;
typedef __attribute__((ext_vector_type(8)))  float  v8f;

static constexpr int Bn  = 8;
static constexpr int LDq = 8192;   // Ld
static constexpr int LMm = 1024;   // Lm
static constexpr int Dd  = 128;    // D == HID
static constexpr float DOT_SCALE = 0.08838834764831845f; // 1/sqrt(128)

// ---- d_out layout (floats, concatenated return tuple) ----
static constexpr size_t OUT_OFF   = 0;                                  // [B][Ld][512]
static constexpr size_t MEM2_OFF  = (size_t)Bn * LDq * 512;             // [B][Lm][128]
static constexpr size_t CROSS_OFF = MEM2_OFF + (size_t)Bn * LMm * Dd;   // [B][Ld][Lm]

// ---- workspace layout (bytes) ----
static constexpr size_t WS_IN_HI  = 0;
static constexpr size_t WS_IN_LO  = WS_IN_HI  + (size_t)Bn*LDq*Dd*2;
static constexpr size_t WS_MEM_HI = WS_IN_LO  + (size_t)Bn*LDq*Dd*2;
static constexpr size_t WS_MEM_LO = WS_MEM_HI + (size_t)Bn*LMm*Dd*2;
static constexpr size_t WS_MEM2T  = WS_MEM_LO + (size_t)Bn*LMm*Dd*2;   // bf16 [B][128][Lm]
static constexpr size_t WS_OFFL   = WS_MEM2T  + (size_t)Bn*Dd*LMm*2;   // f32 [B][Ld]
static constexpr size_t WS_MDOT   = WS_OFFL   + (size_t)Bn*LDq*4;      // f32 [B][Lm]
static constexpr size_t WS_CMAX   = WS_MDOT   + (size_t)Bn*LMm*4;      // u32 enc [B][Lm]
static constexpr size_t WS_CSUM   = WS_CMAX   + (size_t)Bn*LMm*4;      // f32 [B][Lm]
static constexpr size_t WS_OFFM   = WS_CSUM   + (size_t)Bn*LMm*4;      // f32 [B][Lm]
static constexpr size_t WS_CINV   = WS_OFFM   + (size_t)Bn*LMm*4;      // f32 [B][Lm]
static constexpr size_t WS_RMAX   = WS_CINV   + (size_t)Bn*LMm*4;      // u32 enc [B][Ld]
static constexpr size_t WS_W2     = WS_RMAX   + (size_t)Bn*LDq*4;      // f32 [B][Ld]
static constexpr size_t WS_OUT2   = WS_W2     + (size_t)Bn*LDq*4;      // f32 [B][128]

// monotone float<->uint encoding for atomicMax on floats (handles negatives)
__device__ __forceinline__ unsigned fenc(float f) {
    unsigned b = __float_as_uint(f);
    return (b & 0x80000000u) ? ~b : (b | 0x80000000u);
}
__device__ __forceinline__ float fdec(unsigned e) {
    unsigned b = (e & 0x80000000u) ? (e & 0x7fffffffu) : ~e;
    return __uint_as_float(b);
}

__device__ __forceinline__ v8f wmma_bf16(v16bf a, v16bf b, v8f c) {
    return __builtin_amdgcn_wmma_f32_16x16x32_bf16(false, a, false, b, (short)0, c,
                                                   false, false);
}

// A fragment: 16x32 bf16, row-major source, base already offset to (row0, k0)
__device__ __forceinline__ v16bf load_a16(const __bf16* base, int ld, int lane) {
    int row = lane & 15;
    int kb  = (lane & 16) ? 8 : 0;
    const __bf16* p = base + (size_t)row * ld;
    v8bf x = *(const v8bf*)(p + kb);         // K = kb .. kb+7
    v8bf y = *(const v8bf*)(p + 16 + kb);    // K = 16+kb .. 16+kb+7
    return __builtin_shufflevector(x, y, 0,1,2,3,4,5,6,7,8,9,10,11,12,13,14,15);
}
// B fragment: 32x16 bf16; column n of B = row (col0+n) of row-major source.
// base already offset to (col-row0, k0). Lane: col=lane&15, K=kb+e (e<8), 16+kb+(e-8) (e>=8).
__device__ __forceinline__ v16bf load_b16(const __bf16* base, int ld, int lane) {
    int col  = lane & 15;
    int kb   = (lane & 16) ? 8 : 0;
    const __bf16* p = base + (size_t)col * ld;
    v8bf x = *(const v8bf*)(p + kb);
    v8bf y = *(const v8bf*)(p + 16 + kb);
    return __builtin_shufflevector(x, y, 0,1,2,3,4,5,6,7,8,9,10,11,12,13,14,15);
}

// ---------------- kernel 0: init accumulators ----------------
__global__ __launch_bounds__(256) void k_init(unsigned* rmax, unsigned* cmax, float* csum) {
    int i = blockIdx.x * 256 + threadIdx.x;
    unsigned negenc = fenc(-INFINITY);
    if (i < Bn * LDq) rmax[i] = negenc;
    if (i < Bn * LMm) { cmax[i] = negenc; csum[i] = 0.0f; }
}

// ---------------- kernel 1: memory-side prep ----------------
__global__ __launch_bounds__(128) void k_prep_mem(const float* __restrict__ memory,
                                                  const float* __restrict__ w_mem1,
                                                  const float* __restrict__ w_mem2,
                                                  const float* __restrict__ b_mem2,
                                                  float* __restrict__ outp,
                                                  __bf16* mem_hi, __bf16* mem_lo,
                                                  __bf16* mem2T, float* mdot) {
    __shared__ float red[128];
    int t   = threadIdx.x;
    int row = blockIdx.x;            // n*LM + m
    int n = row / LMm, m = row % LMm;
    const float* x = memory + (size_t)row * Dd;

    float xv = x[t];
    __bf16 hi = (__bf16)xv;
    mem_hi[(size_t)row * Dd + t] = hi;
    mem_lo[(size_t)row * Dd + t] = (__bf16)(xv - (float)hi);

    red[t] = xv * w_mem1[t];
    __syncthreads();
    for (int s = 64; s > 0; s >>= 1) { if (t < s) red[t] += red[t + s]; __syncthreads(); }
    if (t == 0) mdot[row] = red[0];

    float acc = b_mem2[t];
    const float* w = w_mem2 + (size_t)t * Dd;
    #pragma unroll 8
    for (int d = 0; d < Dd; ++d) acc = fmaf(x[d], w[d], acc);
    outp[MEM2_OFF + (size_t)row * Dd + t] = acc;                       // mem2 (returned)
    mem2T[((size_t)n * Dd + t) * LMm + m] = (__bf16)acc;               // bf16, transposed
}

// ---------------- kernel 2: input-side prep ----------------
__global__ __launch_bounds__(128) void k_prep_in(const float* __restrict__ input,
                                                 const float* __restrict__ total_mask,
                                                 const float* __restrict__ w_in1,
                                                 const float* __restrict__ w_in2,
                                                 const float* __restrict__ b_in2,
                                                 float* __restrict__ outp,
                                                 __bf16* in_hi, __bf16* in_lo,
                                                 float* offL) {
    __shared__ float red[128];
    int t = threadIdx.x;
    for (int rr = 0; rr < 16; ++rr) {
        size_t row = (size_t)blockIdx.x * 16 + rr;    // n*Ld + l
        const float* x = input + row * Dd;

        float xv = x[t];
        __bf16 hi = (__bf16)xv;
        in_hi[row * Dd + t] = hi;
        in_lo[row * Dd + t] = (__bf16)(xv - (float)hi);

        red[t] = xv * w_in1[t];
        __syncthreads();
        for (int s = 64; s > 0; s >>= 1) { if (t < s) red[t] += red[t + s]; __syncthreads(); }
        if (t == 0) offL[row] = red[0] - 1e30f * total_mask[row];

        float acc = b_in2[t];
        const float* w = w_in2 + (size_t)t * Dd;
        #pragma unroll 8
        for (int d = 0; d < Dd; ++d) acc = fmaf(x[d], w[d], acc);
        outp[OUT_OFF + row * 512 + t] = acc;          // inp2 segment of `out`
        __syncthreads();
    }
}

// ---------------- kernel 3: cross_dot GEMM (bf16x3 WMMA) + att stats ----------------
__global__ __launch_bounds__(128) void k_cross(const __bf16* __restrict__ in_hi,
                                               const __bf16* __restrict__ in_lo,
                                               const __bf16* __restrict__ mem_hi,
                                               const __bf16* __restrict__ mem_lo,
                                               const float* __restrict__ offL,
                                               const float* __restrict__ mdot,
                                               const float* __restrict__ mask,
                                               float* __restrict__ outp,
                                               unsigned* cmax, unsigned* rmax) {
    int lane = threadIdx.x & 31;
    int wave = threadIdx.x >> 5;
    int n    = blockIdx.x >> 7;
    int lblk = blockIdx.x & 127;
    int l0 = lblk * 64 + (wave >> 1) * 32;
    int m0 = blockIdx.y * 64 + (wave & 1) * 32;

    v8f C[2][2];
    v8f z = {0.f,0.f,0.f,0.f,0.f,0.f,0.f,0.f};
    C[0][0] = z; C[0][1] = z; C[1][0] = z; C[1][1] = z;

    #pragma unroll
    for (int kc = 0; kc < 4; ++kc) {
        int k0 = kc * 32;
        v16bf ah[2], al[2], bh[2], bl[2];
        #pragma unroll
        for (int i = 0; i < 2; ++i) {
            size_t abase = ((size_t)(n * LDq + l0 + 16 * i)) * Dd + k0;
            ah[i] = load_a16(in_hi + abase, Dd, lane);
            al[i] = load_a16(in_lo + abase, Dd, lane);
        }
        #pragma unroll
        for (int j = 0; j < 2; ++j) {
            size_t bbase = ((size_t)(n * LMm + m0 + 16 * j)) * Dd + k0;
            bh[j] = load_b16(mem_hi + bbase, Dd, lane);
            bl[j] = load_b16(mem_lo + bbase, Dd, lane);
        }
        #pragma unroll
        for (int i = 0; i < 2; ++i)
            #pragma unroll
            for (int j = 0; j < 2; ++j) {
                C[i][j] = wmma_bf16(ah[i], bh[j], C[i][j]);  // hi*hi
                C[i][j] = wmma_bf16(ah[i], bl[j], C[i][j]);  // hi*lo
                C[i][j] = wmma_bf16(al[i], bh[j], C[i][j]);  // lo*hi
            }
    }

    float* cross = outp + CROSS_OFF;
    #pragma unroll
    for (int i = 0; i < 2; ++i) {
        int lrowbase = l0 + 16 * i + ((lane & 16) ? 8 : 0);
        float rowm[8];
        #pragma unroll
        for (int r = 0; r < 8; ++r) rowm[r] = -INFINITY;

        #pragma unroll
        for (int j = 0; j < 2; ++j) {
            int m  = m0 + 16 * j + (lane & 15);
            int nm = n * LMm + m;
            float moff = mdot[nm] - 1e30f * (1.0f - mask[nm]);
            float colm = -INFINITY;
            #pragma unroll
            for (int r = 0; r < 8; ++r) {
                float cv = C[i][j][r] * DOT_SCALE;
                int   l  = lrowbase + r;
                cross[((size_t)(n * LDq + l)) * LMm + m] = cv;
                float av = cv + offL[n * LDq + l] + moff;
                colm = fmaxf(colm, av);
                rowm[r] = fmaxf(rowm[r], av);
            }
            colm = fmaxf(colm, __shfl_xor(colm, 16, 32));
            if (lane < 16) atomicMax(&cmax[nm], fenc(colm));
        }
        #pragma unroll
        for (int r = 0; r < 8; ++r) {
            float rv = rowm[r];
            rv = fmaxf(rv, __shfl_xor(rv, 1, 32));
            rv = fmaxf(rv, __shfl_xor(rv, 2, 32));
            rv = fmaxf(rv, __shfl_xor(rv, 4, 32));
            rv = fmaxf(rv, __shfl_xor(rv, 8, 32));
            if ((lane & 15) == 0) atomicMax(&rmax[n * LDq + lrowbase + r], fenc(rv));
        }
    }
}

// ---------------- kernel 4: column sum of exp(att - cmax) ----------------
__global__ __launch_bounds__(256) void k_colsum(const float* __restrict__ outp,
                                                const float* __restrict__ offL,
                                                const float* __restrict__ mdot,
                                                const float* __restrict__ mask,
                                                const unsigned* __restrict__ cmax,
                                                float* csum) {
    __shared__ float red[256];
    int t  = threadIdx.x;
    int mt = t & 63, lt = t >> 6;
    int n = blockIdx.x >> 4, lchunk = blockIdx.x & 15;
    int m  = blockIdx.y * 64 + mt;
    int nm = n * LMm + m;
    float moff = mdot[nm] - 1e30f * (1.0f - mask[nm]) - fdec(cmax[nm]);
    const float* cross = outp + CROSS_OFF + ((size_t)n * LDq) * LMm;

    float s = 0.0f;
    #pragma unroll 4
    for (int i = 0; i < 128; ++i) {
        int l = lchunk * 512 + lt + 4 * i;
        s += __expf(cross[(size_t)l * LMm + m] + offL[n * LDq + l] + moff);
    }
    red[t] = s;
    __syncthreads();
    if (lt == 0) {
        float tot = red[mt] + red[64 + mt] + red[128 + mt] + red[192 + mt];
        atomicAdd(&csum[nm], tot);
    }
}

// ---------------- kernel 5: fold column stats ----------------
__global__ __launch_bounds__(256) void k_fincol(const float* __restrict__ mdot,
                                                const float* __restrict__ mask,
                                                const unsigned* __restrict__ cmax,
                                                const float* __restrict__ csum,
                                                float* offM, float* cinv) {
    int i = blockIdx.x * 256 + threadIdx.x;
    if (i >= Bn * LMm) return;
    offM[i] = mdot[i] - 1e30f * (1.0f - mask[i]) - fdec(cmax[i]);
    float cs = csum[i];
    cinv[i] = (cs > 0.0f) ? (mask[i] / cs) : 0.0f;
}

// ---------------- kernel 6: weight_two = softmax over l of rowmax ----------------
__global__ __launch_bounds__(256) void k_rowsoft(const unsigned* __restrict__ rmax,
                                                 float* w2) {
    __shared__ float red[256];
    __shared__ float sM, sS;
    int t = threadIdx.x, n = blockIdx.x;
    const unsigned* rm = rmax + (size_t)n * LDq;

    float mx = -INFINITY;
    for (int l = t; l < LDq; l += 256) mx = fmaxf(mx, fdec(rm[l]));
    red[t] = mx; __syncthreads();
    for (int s = 128; s > 0; s >>= 1) { if (t < s) red[t] = fmaxf(red[t], red[t + s]); __syncthreads(); }
    if (t == 0) sM = red[0];
    __syncthreads();

    float sum = 0.0f;
    for (int l = t; l < LDq; l += 256) sum += __expf(fdec(rm[l]) - sM);
    red[t] = sum; __syncthreads();
    for (int s = 128; s > 0; s >>= 1) { if (t < s) red[t] += red[t + s]; __syncthreads(); }
    if (t == 0) sS = red[0];
    __syncthreads();

    float inv = 1.0f / sS;
    for (int l = t; l < LDq; l += 256)
        w2[(size_t)n * LDq + l] = __expf(fdec(rm[l]) - sM) * inv;
}

// ---------------- kernel 7: output_two = sum_l w2[l]*inp2[l,h] ----------------
__global__ __launch_bounds__(128) void k_out2(const float* __restrict__ outp,
                                              const float* __restrict__ w2,
                                              float* out2) {
    int h = threadIdx.x, n = blockIdx.x;
    const float* inp2 = outp + OUT_OFF + ((size_t)n * LDq) * 512;
    float acc = 0.0f;
    for (int l = 0; l < LDq; ++l)
        acc = fmaf(w2[(size_t)n * LDq + l], inp2[(size_t)l * 512 + h], acc);
    out2[n * Dd + h] = acc;
}

// ---- async copy: 64B of one mem2T row -> LDS (ASYNCcnt path, §08 async_tensor) ----
__device__ __forceinline__ void stage_row_async(const __bf16* grow, unsigned lds_byte) {
    asm volatile(
        "global_load_async_to_lds_b128 %0, %1, off\n\t"
        "global_load_async_to_lds_b128 %0, %1, off offset:16\n\t"
        "global_load_async_to_lds_b128 %0, %1, off offset:32\n\t"
        "global_load_async_to_lds_b128 %0, %1, off offset:48"
        :: "v"(lds_byte), "v"(grow) : "memory");
}
__device__ __forceinline__ void wait_async0() {
    asm volatile("s_wait_asynccnt 0x0" ::: "memory");
}

// ---------------- kernel 8: output_one GEMM (WMMA) + fused concat epilogue ----------
// block = 4 waves; each wave: 16(l) x 128(h) C macro-tile, K = m = 1024 in 32-chunks.
// A (weight_one) rebuilt on the fly from cross_dot (vectorized f32 stream);
// B chunk (128h x 32m bf16 = 8KB) staged cooperatively into LDS with
// global_load_async_to_lds_b128, double-buffered across the K loop.
__global__ __launch_bounds__(128) void k_pass3(const float* __restrict__ offL,
                                               const float* __restrict__ offM,
                                               const float* __restrict__ cinv,
                                               const __bf16* __restrict__ mem2T,
                                               const float* __restrict__ out2,
                                               float* __restrict__ outp) {
    __shared__ __align__(64) __bf16 sB[2][128 * 32];   // 2 x 8KB double buffer
    int t    = threadIdx.x;
    int lane = t & 31;
    int wave = t >> 5;
    int n    = blockIdx.x >> 7;
    int lblk = blockIdx.x & 127;
    int ltile = lblk * 64 + wave * 16;

    int arow = ltile + (lane & 15);
    const float* crow  = outp + CROSS_OFF + ((size_t)(n * LDq + arow)) * LMm;
    float offLl        = offL[n * LDq + arow];
    const float* offMn = offM + (size_t)n * LMm;
    const float* cinvn = cinv + (size_t)n * LMm;
    const __bf16* m2n  = mem2T + (size_t)n * Dd * LMm;   // [128][1024]

    unsigned sb0 = (unsigned)(uintptr_t)&sB[0][0];       // LDS byte offset (addr[31:0])
    unsigned sb1 = (unsigned)(uintptr_t)&sB[1][0];

    v8f C[8];
    v8f z = {0.f,0.f,0.f,0.f,0.f,0.f,0.f,0.f};
    #pragma unroll
    for (int j = 0; j < 8; ++j) C[j] = z;

    int kb   = (lane & 16) ? 8 : 0;
    int koff = (lane & 16) ? 16 : 0;

    // prologue: stage chunk 0 (thread t stages row h=t, 64 bytes)
    stage_row_async(m2n + (size_t)t * LMm + 0, sb0 + (unsigned)t * 64u);

    for (int mc = 0; mc < 32; ++mc) {
        int m0 = mc * 32;
        wait_async0();
        __syncthreads();                      // chunk mc resident; buf^1 free to restage
        if (mc < 31)
            stage_row_async(m2n + (size_t)t * LMm + (m0 + 32),
                            ((mc & 1) ? sb0 : sb1) + (unsigned)t * 64u);

        // rebuild weight_one fragment from cross_dot (aligned float4 streams)
        float cv[16], ov[16], iv[16];
        *(float4*)(cv + 0)  = *(const float4*)(crow  + m0 + kb);
        *(float4*)(cv + 4)  = *(const float4*)(crow  + m0 + kb + 4);
        *(float4*)(cv + 8)  = *(const float4*)(crow  + m0 + 16 + kb);
        *(float4*)(cv + 12) = *(const float4*)(crow  + m0 + 16 + kb + 4);
        *(float4*)(ov + 0)  = *(const float4*)(offMn + m0 + kb);
        *(float4*)(ov + 4)  = *(const float4*)(offMn + m0 + kb + 4);
        *(float4*)(ov + 8)  = *(const float4*)(offMn + m0 + 16 + kb);
        *(float4*)(ov + 12) = *(const float4*)(offMn + m0 + 16 + kb + 4);
        *(float4*)(iv + 0)  = *(const float4*)(cinvn + m0 + kb);
        *(float4*)(iv + 4)  = *(const float4*)(cinvn + m0 + kb + 4);
        *(float4*)(iv + 8)  = *(const float4*)(cinvn + m0 + 16 + kb);
        *(float4*)(iv + 12) = *(const float4*)(cinvn + m0 + 16 + kb + 4);
        __builtin_prefetch(crow + m0 + 64, 0, 1);

        v16bf a;
        #pragma unroll
        for (int e = 0; e < 16; ++e)
            a[e] = (__bf16)(__expf(cv[e] + offLl + ov[e]) * iv[e]);

        const __bf16* sbuf = sB[mc & 1];
        #pragma unroll
        for (int j = 0; j < 8; ++j) {
            const __bf16* bp = sbuf + (j * 16 + (lane & 15)) * 32;
            v8bf bx = *(const v8bf*)(bp + kb);
            v8bf by = *(const v8bf*)(bp + 16 + kb);
            v16bf b = __builtin_shufflevector(bx, by, 0,1,2,3,4,5,6,7,8,9,10,11,12,13,14,15);
            C[j] = wmma_bf16(a, b, C[j]);
        }
    }

    // fused concat: out = [inp2 | output_one | inp2*o1 | out2*o1]
    int rowbase = ltile + ((lane & 16) ? 8 : 0);
    #pragma unroll
    for (int j = 0; j < 8; ++j) {
        int h = j * 16 + (lane & 15);
        float o2 = out2[n * Dd + h];
        #pragma unroll
        for (int r = 0; r < 8; ++r) {
            size_t ob = ((size_t)(n * LDq + rowbase + r)) * 512 + h;
            float o1 = C[j][r];
            float i2 = outp[OUT_OFF + ob];
            outp[OUT_OFF + ob + 128] = o1;
            outp[OUT_OFF + ob + 256] = i2 * o1;
            outp[OUT_OFF + ob + 384] = o2 * o1;
        }
    }
}

extern "C" void kernel_launch(void* const* d_in, const int* in_sizes, int n_in,
                              void* d_out, int out_size, void* d_ws, size_t ws_size,
                              hipStream_t stream) {
    (void)in_sizes; (void)n_in; (void)out_size; (void)ws_size;
    const float* input      = (const float*)d_in[0];
    const float* memory     = (const float*)d_in[1];
    const float* mask       = (const float*)d_in[2];
    const float* total_mask = (const float*)d_in[3];
    const float* w_in1      = (const float*)d_in[4];
    const float* w_mem1     = (const float*)d_in[5];
    const float* w_in2      = (const float*)d_in[6];
    const float* b_in2      = (const float*)d_in[7];
    const float* w_mem2     = (const float*)d_in[8];
    const float* b_mem2     = (const float*)d_in[9];
    float* outp = (float*)d_out;

    char* ws = (char*)d_ws;
    __bf16*   in_hi  = (__bf16*)(ws + WS_IN_HI);
    __bf16*   in_lo  = (__bf16*)(ws + WS_IN_LO);
    __bf16*   mem_hi = (__bf16*)(ws + WS_MEM_HI);
    __bf16*   mem_lo = (__bf16*)(ws + WS_MEM_LO);
    __bf16*   mem2T  = (__bf16*)(ws + WS_MEM2T);
    float*    offL   = (float*)  (ws + WS_OFFL);
    float*    mdot   = (float*)  (ws + WS_MDOT);
    unsigned* cmax   = (unsigned*)(ws + WS_CMAX);
    float*    csum   = (float*)  (ws + WS_CSUM);
    float*    offM   = (float*)  (ws + WS_OFFM);
    float*    cinv   = (float*)  (ws + WS_CINV);
    unsigned* rmax   = (unsigned*)(ws + WS_RMAX);
    float*    w2     = (float*)  (ws + WS_W2);
    float*    out2   = (float*)  (ws + WS_OUT2);

    k_init<<<(Bn * LDq + 255) / 256, 256, 0, stream>>>(rmax, cmax, csum);
    k_prep_mem<<<Bn * LMm, 128, 0, stream>>>(memory, w_mem1, w_mem2, b_mem2, outp,
                                             mem_hi, mem_lo, mem2T, mdot);
    k_prep_in<<<Bn * LDq / 16, 128, 0, stream>>>(input, total_mask, w_in1, w_in2, b_in2,
                                                 outp, in_hi, in_lo, offL);
    k_cross<<<dim3(Bn * (LDq / 64), LMm / 64), 128, 0, stream>>>(
        in_hi, in_lo, mem_hi, mem_lo, offL, mdot, mask, outp, cmax, rmax);
    k_colsum<<<dim3(Bn * (LDq / 512), LMm / 64), 256, 0, stream>>>(
        outp, offL, mdot, mask, cmax, csum);
    k_fincol<<<(Bn * LMm + 255) / 256, 256, 0, stream>>>(mdot, mask, cmax, csum, offM, cinv);
    k_rowsoft<<<Bn, 256, 0, stream>>>(rmax, w2);
    k_out2<<<Bn, 128, 0, stream>>>(outp, w2, out2);
    k_pass3<<<Bn * (LDq / 64), 128, 0, stream>>>(offL, offM, cinv, mem2T, out2, outp);
}
